// GCU_41832981463347
// MI455X (gfx1250) — matrix operand order repaired and checked by reading, not compiled, not run on hardware
//
#include <hip/hip_runtime.h>
#include <hip/hip_fp16.h>

// MI455X (gfx1250, wave32). D=256, H=W=512 -> P=262144 rows, V=32, OUTF=256.
// Memory-bound (~0.9 GB moved -> ~39us at 23.3 TB/s); all big GEMMs use
// v_wmma_f32_16x16x32_f16 (f16 in / f32 accumulate).

typedef __attribute__((ext_vector_type(16))) _Float16 v16h;
typedef __attribute__((ext_vector_type(8)))  _Float16 v8h;
typedef __attribute__((ext_vector_type(8)))  float    v8f;

static constexpr int DDIM = 256;
static constexpr int VDIM = 32;
static constexpr int OUTD = 256;
static constexpr int P_TOTAL = 512 * 512;        // 262144
static constexpr int K2_BLOCKS = P_TOTAL / 256;  // 1024 (256 rows / WG, 2 tiles/wave)
static constexpr int K7_BLOCKS = P_TOTAL / 128;  // 2048 (128 rows / WG)
static constexpr int K4_BLOCKS = 128;            // split-K parts for XtQ

// ---------------------------------------------------------------------------
// k1: precompute bias + pre-swizzled f16 B operands for the Q GEMM.
// B-matrix (32x16) per-lane layout: lane L holds column n=L%16,
// element e holds K = 8*(L/16) + (e&7) + 16*(e>>3).  (ISA 7.12.2)
// ---------------------------------------------------------------------------
__global__ void k1_precompute(const float* __restrict__ W,
                              const float* __restrict__ var,
                              _Float16* __restrict__ Bw,   // -2*W*iv2, swizzled
                              _Float16* __restrict__ Bi,   // iv2, swizzled
                              float* __restrict__ bbias,   // [32]
                              float* __restrict__ iv_f32)  // 1/var, [256][32]
{
  const int t = threadIdx.x; // 256 threads
  for (int i = 0; i < VDIM; ++i)
    iv_f32[t * VDIM + i] = 1.0f / var[t * VDIM + i];

  if (t < VDIM) {
    float acc = 0.f;
    for (int d = 0; d < DDIM; ++d) {
      float w = W[d * VDIM + t];
      float v = var[d * VDIM + t];
      float iv2 = 1.0f / (v * v);
      acc += w * w * iv2;
    }
    bbias[t] = acc;
  }
  // swizzled layout index: ((kc*2+nt)*32 + lane)*16 + e
  for (int j = 0; j < 32; ++j) {
    int f    = t * 32 + j;          // 0..8191
    int e    = f & 15;
    int lane = (f >> 4) & 31;
    int nt   = (f >> 9) & 1;
    int kc   = f >> 10;
    int d    = kc * 32 + (e & 7) + 8 * (lane >> 4) + 16 * (e >> 3);   // B K-index -> d
    int i    = nt * 16 + (lane & 15);            // column -> V index
    float v  = var[d * VDIM + i];
    float iv2 = 1.0f / (v * v);
    Bw[f] = (_Float16)(-2.0f * W[d * VDIM + i] * iv2);
    Bi[f] = (_Float16)(iv2);
  }
}

// ---------------------------------------------------------------------------
// k2: fused Q pass.  One wave = two 16-row tiles (B operands amortized over
// both); 8 waves/WG = 256 rows.  The x- and x^2-GEMMs are chained into ONE
// accumulator per output tile (D->C accumulation, 4 independent chains),
// halving accumulator VGPRs vs separate accumulators (no spills).
// A-matrix (16x32) per-lane layout: lane L holds row m=L%16; elements
// e=0..7 -> K = 8*(L/16)+e,  e=8..15 -> K = 16+8*(L/16)+(e-8).  (ISA 7.12.2)
// ---------------------------------------------------------------------------
__global__ void k2_q(const float* __restrict__ X,
                     const _Float16* __restrict__ Bw,
                     const _Float16* __restrict__ Bi,
                     const float* __restrict__ bbias,
                     float* __restrict__ Q,
                     float* __restrict__ s_part)
{
  __shared__ float s_lds[8][VDIM];
  const int t    = threadIdx.x;
  const int wv   = t >> 5;
  const int lane = t & 31;
  const int hf   = lane >> 4;
  const int lm   = lane & 15;
  const int p0   = blockIdx.x * 256 + wv * 32;   // wave covers rows p0..p0+31
  const float* xrow0 = X + (size_t)(p0 + lm) * DDIM;
  const float* xrow1 = xrow0 + (size_t)16 * DDIM;

  v8f acc[2][2] = {};   // [tile][nt]; accumulates BOTH x and x^2 products
#pragma unroll
  for (int kc = 0; kc < 8; ++kc) {
    const int g1 = kc * 32 + 8 * hf;
    const int g2 = g1 + 16;
    v16h bw0 = *(const v16h*)(Bw + ((kc * 2 + 0) * 32 + lane) * 16);
    v16h bi0 = *(const v16h*)(Bi + ((kc * 2 + 0) * 32 + lane) * 16);
    v16h bw1 = *(const v16h*)(Bw + ((kc * 2 + 1) * 32 + lane) * 16);
    v16h bi1 = *(const v16h*)(Bi + ((kc * 2 + 1) * 32 + lane) * 16);
#pragma unroll
    for (int tl = 0; tl < 2; ++tl) {
      const float* xr = tl ? xrow1 : xrow0;
      float f[16];
      *(float4*)(f + 0)  = *(const float4*)(xr + g1);
      *(float4*)(f + 4)  = *(const float4*)(xr + g1 + 4);
      *(float4*)(f + 8)  = *(const float4*)(xr + g2);
      *(float4*)(f + 12) = *(const float4*)(xr + g2 + 4);
      v16h a, asq;
#pragma unroll
      for (int e = 0; e < 16; ++e) {
        a[e]   = (_Float16)f[e];
        asq[e] = (_Float16)(f[e] * f[e]);
      }
      acc[tl][0] = __builtin_amdgcn_wmma_f32_16x16x32_f16(false, a,   false, bw0, (short)0, acc[tl][0], false, false);
      acc[tl][0] = __builtin_amdgcn_wmma_f32_16x16x32_f16(false, asq, false, bi0, (short)0, acc[tl][0], false, false);
      acc[tl][1] = __builtin_amdgcn_wmma_f32_16x16x32_f16(false, a,   false, bw1, (short)0, acc[tl][1], false, false);
      acc[tl][1] = __builtin_amdgcn_wmma_f32_16x16x32_f16(false, asq, false, bi1, (short)0, acc[tl][1], false, false);
    }
  }

  const float b0 = bbias[lm];
  const float b1 = bbias[16 + lm];
  float cs0 = 0.f, cs1 = 0.f;
#pragma unroll
  for (int tl = 0; tl < 2; ++tl) {
#pragma unroll
    for (int r = 0; r < 8; ++r) {
      float q0 = acc[tl][0][r] + b0;  // column lm
      float q1 = acc[tl][1][r] + b1;  // column 16+lm
      // row = p0 + 16*tl + r + 8*hf ; reduce over 32 columns
      float mn = fminf(q0, q1);
      mn = fminf(mn, __shfl_xor(mn, 1, 32));
      mn = fminf(mn, __shfl_xor(mn, 2, 32));
      mn = fminf(mn, __shfl_xor(mn, 4, 32));
      mn = fminf(mn, __shfl_xor(mn, 8, 32));
      float e0 = __expf(-0.5f * (q0 - mn));
      float e1 = __expf(-0.5f * (q1 - mn));
      float rs = e0 + e1;
      rs += __shfl_xor(rs, 1, 32);
      rs += __shfl_xor(rs, 2, 32);
      rs += __shfl_xor(rs, 4, 32);
      rs += __shfl_xor(rs, 8, 32);
      float inv = 1.0f / rs;
      e0 *= inv; e1 *= inv;
      const size_t prow = (size_t)(p0 + 16 * tl + r + 8 * hf);
      Q[prow * VDIM + lm]      = e0;
      Q[prow * VDIM + 16 + lm] = e1;
      cs0 += e0; cs1 += e1;
    }
  }
  // column sums: lane l and l+16 hold same column, rows 0-7 vs 8-15
  cs0 += __shfl_xor(cs0, 16, 32);
  cs1 += __shfl_xor(cs1, 16, 32);
  if (lane < 16) { s_lds[wv][lm] = cs0; s_lds[wv][16 + lm] = cs1; }
  __syncthreads();
  if (t < VDIM) {
    float acc2 = 0.f;
#pragma unroll
    for (int w = 0; w < 8; ++w) acc2 += s_lds[w][t];
    s_part[(size_t)blockIdx.x * VDIM + t] = acc2;  // fixed order: deterministic
  }
}

// ---------------------------------------------------------------------------
// k3: s = sum of per-WG partials (fixed order), plus 1/s.
// ---------------------------------------------------------------------------
__global__ void k3_reduce_s(const float* __restrict__ s_part,
                            float* __restrict__ s, float* __restrict__ inv_s)
{
  const int t = threadIdx.x; // 32
  if (t < VDIM) {
    float acc = 0.f;
    for (int w = 0; w < K2_BLOCKS; ++w) acc += s_part[(size_t)w * VDIM + t];
    s[t] = acc;
    inv_s[t] = 1.0f / acc;
  }
}

// ---------------------------------------------------------------------------
// k4: XtQ = X^T @ Q, split-K over 128 WGs (2048 rows each).
// Stage 64-row chunks of X and Q into LDS transposed as f16, then WMMA with
// A from lds_xt (d-major) and B from lds_qt (v-major).  Software prefetch of
// the next chunk (global_prefetch_b8) overlaps staging with HBM latency.
// ---------------------------------------------------------------------------
__global__ void k4_xtq(const float* __restrict__ X,
                       const float* __restrict__ Q,
                       float* __restrict__ part)
{
  __shared__ _Float16 xt[DDIM][64];  // 32 KB, X^T chunk
  __shared__ _Float16 qt[VDIM][64];  //  4 KB, Q^T chunk
  const int t    = threadIdx.x;
  const int wv   = t >> 5;
  const int lane = t & 31;
  const int hf   = lane >> 4;
  const int lm   = lane & 15;
  const size_t pbase = (size_t)blockIdx.x * 2048;

  v8f acc[2][2] = {};
  for (int chunk = 0; chunk < 32; ++chunk) {
    const size_t p0 = pbase + (size_t)chunk * 64;
    __syncthreads();
    {
      const int pr = t & 63;
      const int cg = t >> 6; // 0..3
      const float* row = X + (p0 + pr) * DDIM + cg * 64;
      if (chunk + 1 < 32) {  // prefetch next chunk's X rows into caches
        __builtin_prefetch(row + (size_t)64 * DDIM, 0, 3);
      }
#pragma unroll
      for (int q4 = 0; q4 < 16; ++q4) {
        float4 v = *(const float4*)(row + q4 * 4);
        int c = cg * 64 + q4 * 4;
        xt[c + 0][pr] = (_Float16)v.x;
        xt[c + 1][pr] = (_Float16)v.y;
        xt[c + 2][pr] = (_Float16)v.z;
        xt[c + 3][pr] = (_Float16)v.w;
      }
      const float* qrow = Q + (p0 + pr) * VDIM + cg * 8;
#pragma unroll
      for (int q4 = 0; q4 < 2; ++q4) {
        float4 v = *(const float4*)(qrow + q4 * 4);
        int c = cg * 8 + q4 * 4;
        qt[c + 0][pr] = (_Float16)v.x;
        qt[c + 1][pr] = (_Float16)v.y;
        qt[c + 2][pr] = (_Float16)v.z;
        qt[c + 3][pr] = (_Float16)v.w;
      }
    }
    __syncthreads();
#pragma unroll
    for (int kb = 0; kb < 64; kb += 32) {
#pragma unroll
      for (int mtl = 0; mtl < 2; ++mtl) {
        const int d = (wv * 2 + mtl) * 16 + lm;      // A row (d index)
        v8h a0 = *(const v8h*)(&xt[d][kb + 16 * hf]);
        v8h a1 = *(const v8h*)(&xt[d][kb + 16 * hf + 8]);
        v16h a;
#pragma unroll
        for (int e = 0; e < 8; ++e) { a[e] = a0[e]; a[8 + e] = a1[e]; }
#pragma unroll
        for (int nt = 0; nt < 2; ++nt) {
          const int vcol = nt * 16 + lm;             // B column (V index)
          v16h b = *(const v16h*)(&qt[vcol][kb + 16 * hf]);
          acc[mtl][nt] = __builtin_amdgcn_wmma_f32_16x16x32_f16(
              false, a, false, b, (short)0, acc[mtl][nt], false, false);
        }
      }
    }
  }
  float* dst = part + (size_t)blockIdx.x * (DDIM * VDIM);
#pragma unroll
  for (int mtl = 0; mtl < 2; ++mtl)
#pragma unroll
    for (int nt = 0; nt < 2; ++nt)
#pragma unroll
      for (int r = 0; r < 8; ++r) {
        const int d = (wv * 2 + mtl) * 16 + r + 8 * hf;
        const int i = nt * 16 + lm;
        dst[d * VDIM + i] = acc[mtl][nt][r];
      }
}

// ---------------------------------------------------------------------------
// k5a: reduce XtQ split-K partials (fixed order).  k5b: finalize Z.
// ---------------------------------------------------------------------------
__global__ void k5a_reduce_xtq(const float* __restrict__ part,
                               float* __restrict__ XtQ)
{
  const int o = blockIdx.x * 256 + threadIdx.x; // 0..8191
  float acc = 0.f;
  for (int w = 0; w < K4_BLOCKS; ++w)
    acc += part[(size_t)w * (DDIM * VDIM) + o];
  XtQ[o] = acc;
}

__global__ void k5b_z(const float* __restrict__ XtQ,
                      const float* __restrict__ W,
                      const float* __restrict__ iv,
                      const float* __restrict__ s,
                      const float* __restrict__ inv_s,
                      float* __restrict__ Z)
{
  __shared__ float zl[DDIM * VDIM];
  __shared__ float nrm[VDIM];
  const int d = threadIdx.x; // 256
  for (int i = 0; i < VDIM; ++i) {
    float z = (XtQ[d * VDIM + i] - W[d * VDIM + i] * s[i]) * iv[d * VDIM + i] * inv_s[i];
    zl[d * VDIM + i] = z;
  }
  __syncthreads();
  if (d < VDIM) {
    float acc = 0.f;
    for (int k = 0; k < DDIM; ++k) { float z = zl[k * VDIM + d]; acc += z * z; }
    nrm[d] = 1.0f / acc;  // reference divides by sum of squares (no sqrt)
  }
  __syncthreads();
  for (int i = 0; i < VDIM; ++i)
    Z[d * VDIM + i] = zl[d * VDIM + i] * nrm[i];
}

// ---------------------------------------------------------------------------
// k6: Adj = Z^T Z ; Zo = relu(Adj @ (Z^T @ weight)) ; emit Zo pre-swizzled
// as the f16 B operand (K=32) for k7.
// ---------------------------------------------------------------------------
__global__ void k6_small(const float* __restrict__ Z,
                         const float* __restrict__ weight,
                         _Float16* __restrict__ Bzo)
{
  __shared__ float zl[DDIM * VDIM];   // Z, later reused for Zo
  __shared__ float zw[VDIM * OUTD];   // Z^T @ weight
  __shared__ float adj[VDIM * VDIM];
  const int t = threadIdx.x; // 256
  for (int i = 0; i < VDIM; ++i) zl[t * VDIM + i] = Z[t * VDIM + i];
  __syncthreads();
  for (int k = 0; k < 32; ++k) {           // ZtW: i=k, f=t
    float acc = 0.f;
    for (int d = 0; d < DDIM; ++d) acc += zl[d * VDIM + k] * weight[d * OUTD + t];
    zw[k * OUTD + t] = acc;
  }
  for (int k = 0; k < 4; ++k) {            // Adj
    const int o = t + 256 * k;
    const int i = o >> 5, j = o & 31;
    float acc = 0.f;
    for (int d = 0; d < DDIM; ++d) acc += zl[d * VDIM + i] * zl[d * VDIM + j];
    adj[o] = acc;
  }
  __syncthreads();
  float zo_local[32];
  for (int k = 0; k < 32; ++k) {           // Zo: i=k, f=t
    float acc = 0.f;
    for (int j = 0; j < VDIM; ++j) acc += adj[k * VDIM + j] * zw[j * OUTD + t];
    zo_local[k] = fmaxf(acc, 0.f);
  }
  __syncthreads();
  for (int k = 0; k < 32; ++k) zl[k * OUTD + t] = zo_local[k]; // zl := Zo[i][f]
  __syncthreads();
  for (int j = 0; j < 32; ++j) {           // swizzle -> B operand, K=32
    const int fidx = t * 32 + j;
    const int e    = fidx & 15;
    const int lane = (fidx >> 4) & 31;
    const int nt   = fidx >> 9;            // 0..15 N-tiles
    const int kk   = (e & 7) + 8 * (lane >> 4) + 16 * (e >> 3);
    const int n    = nt * 16 + (lane & 15);
    Bzo[fidx] = (_Float16)zl[kk * OUTD + n];
  }
}

// ---------------------------------------------------------------------------
// k7: Xn = Q @ Zo.  K=32 -> single WMMA per 16x16 tile; 16 N-tiles per wave
// reuse one A operand.  Results staged through a private per-wave LDS region
// so the 268 MB output is written as fully coalesced b128 stores.
// ---------------------------------------------------------------------------
__global__ void k7_out(const float* __restrict__ Q,
                       const _Float16* __restrict__ Bzo,
                       float* __restrict__ out)
{
  __shared__ float obuf[8 * 16 * OUTD];   // 128 KB: 16 rows x 256 cols / wave
  const int t    = threadIdx.x;
  const int wv   = t >> 5;
  const int lane = t & 31;
  const int hf   = lane >> 4;
  const int lm   = lane & 15;
  const size_t p0 = (size_t)blockIdx.x * 128 + (size_t)wv * 16;
  const float* qrow = Q + (p0 + lm) * VDIM;
  float* myo = obuf + wv * 16 * OUTD;

  float f[16];
  const int g1 = 8 * hf;
  *(float4*)(f + 0)  = *(const float4*)(qrow + g1);
  *(float4*)(f + 4)  = *(const float4*)(qrow + g1 + 4);
  *(float4*)(f + 8)  = *(const float4*)(qrow + g1 + 16);
  *(float4*)(f + 12) = *(const float4*)(qrow + g1 + 20);
  v16h a;
#pragma unroll
  for (int e = 0; e < 16; ++e) a[e] = (_Float16)f[e];

#pragma unroll
  for (int nt = 0; nt < 16; ++nt) {
    v16h b = *(const v16h*)(Bzo + (nt * 32 + lane) * 16);
    v8f c = {};
    c = __builtin_amdgcn_wmma_f32_16x16x32_f16(false, a, false, b, (short)0, c, false, false);
#pragma unroll
    for (int r = 0; r < 8; ++r)
      myo[(r + 8 * hf) * OUTD + nt * 16 + lm] = c[r];
  }
  __syncthreads();
  // coalesced writeback: 1024 float4 per wave, 32 per lane
#pragma unroll
  for (int j = 0; j < 32; ++j) {
    const int f4   = j * 32 + lane;      // 0..1023
    const int row  = f4 >> 6;
    const int col4 = (f4 & 63) * 4;
    float4 v = *(const float4*)(myo + row * OUTD + col4);
    *(float4*)(out + (p0 + row) * OUTD + col4) = v;
  }
}

// ---------------------------------------------------------------------------
extern "C" void kernel_launch(void* const* d_in, const int* in_sizes, int n_in,
                              void* d_out, int out_size, void* d_ws, size_t ws_size,
                              hipStream_t stream) {
  const float* X      = (const float*)d_in[0];   // (1,256,512,512) flat
  const float* W      = (const float*)d_in[1];   // (256,32)
  const float* var    = (const float*)d_in[2];   // (256,32)
  const float* weight = (const float*)d_in[3];   // (256,256)
  float* out = (float*)d_out;

  char* ws = (char*)d_ws;
  size_t off = 0;
  auto carve = [&](size_t bytes) -> char* {
    char* p = ws + off;
    off = (off + bytes + 255) & ~(size_t)255;
    return p;
  };
  float*    Q        = (float*)   carve((size_t)P_TOTAL * VDIM * 4);        // 33.5 MB
  _Float16* Bw       = (_Float16*)carve(8192 * 2);
  _Float16* Bi       = (_Float16*)carve(8192 * 2);
  float*    bbias    = (float*)   carve(VDIM * 4);
  float*    iv_f32   = (float*)   carve(8192 * 4);
  float*    s_part   = (float*)   carve((size_t)K2_BLOCKS * VDIM * 4);
  float*    s        = (float*)   carve(VDIM * 4);
  float*    inv_s    = (float*)   carve(VDIM * 4);
  float*    xtq_part = (float*)   carve((size_t)K4_BLOCKS * 8192 * 4);      // 4 MB
  float*    XtQ      = (float*)   carve(8192 * 4);
  float*    Z        = (float*)   carve(8192 * 4);
  _Float16* Bzo      = (_Float16*)carve(8192 * 2);
  (void)ws_size; (void)in_sizes; (void)n_in; (void)out_size;

  k1_precompute<<<1, 256, 0, stream>>>(W, var, Bw, Bi, bbias, iv_f32);
  k2_q<<<K2_BLOCKS, 256, 0, stream>>>(X, Bw, Bi, bbias, Q, s_part);
  k3_reduce_s<<<1, 32, 0, stream>>>(s_part, s, inv_s);
  k4_xtq<<<K4_BLOCKS, 256, 0, stream>>>(X, Q, xtq_part);
  k5a_reduce_xtq<<<32, 256, 0, stream>>>(xtq_part, XtQ);
  k5b_z<<<1, 256, 0, stream>>>(XtQ, W, iv_f32, s, inv_s, Z);
  k6_small<<<1, 256, 0, stream>>>(Z, weight, Bzo);
  k7_out<<<K7_BLOCKS, 256, 0, stream>>>(Q, Bzo, out);
}
